// DKF1_Variational_15762529976857
// MI455X (gfx1250) — matrix operand order repaired and checked by reading, not compiled, not run on hardware
//
#include <hip/hip_runtime.h>
#include <cstdint>
#include <cstddef>

// ---------------------------------------------------------------------------
// DKF variational model, CDNA5 (gfx1250).
// B=4096, L=128, DU=6, H=64 (3H=192), DZ=6, DY=1.
//   * batch-slab persistent kernels; recurrences loop t privately per WG.
//   * all dense math via v_wmma_f32_16x16x32_bf16; weights repacked once per
//     launch into the WMMA B-fragment lane layout (bf16) in workspace.
//   * GEMM C fragments initialized from pre-packed bias vectors (one
//     unconditional load per tile) and stored directly (no LDS RMW).
//   * per-timestep slab copies use global_load_async_to_lds_b128 + ASYNCcnt.
// ---------------------------------------------------------------------------

typedef __attribute__((ext_vector_type(16))) __bf16 v16bf;
typedef __attribute__((ext_vector_type(8)))  float  v8f;

#define NWAVES 8           // 256 threads, wave32
constexpr int Bb = 4096, Lt = 128, DU = 6, H = 64, H3 = 192, DZ = 6;
constexpr size_t BL = (size_t)Bb * Lt;

// --- assumed d_in flattening: top dict insertion order (y_seq, u_seq, params),
// --- params pytree with sorted dict keys, lists/tuples in order.
enum {
  IN_Y = 0, IN_U = 1,
  IN_DEC0_W = 2, IN_DEC0_B = 3, IN_DEC1_W = 4, IN_DEC1_B = 5,
  IN_DECMEAN_W = 6, IN_DECMEAN_B = 7,
  IN_G1B_BI = 8, IN_G1B_BR = 9, IN_G1B_K = 10, IN_G1B_RK = 11,
  IN_G1F_BI = 12, IN_G1F_BR = 13, IN_G1F_K = 14, IN_G1F_RK = 15,
  IN_G2B_BI = 16, IN_G2B_BR = 17, IN_G2B_K = 18, IN_G2B_RK = 19,
  IN_G2F_BI = 20, IN_G2F_BR = 21, IN_G2F_K = 22, IN_G2F_RK = 23,
  IN_POST0_W = 24, IN_POST0_B = 25, IN_POST1_W = 26, IN_POST1_B = 27,
  IN_POSTLV_W = 28, IN_POSTLV_B = 29, IN_POSTMU_W = 30, IN_POSTMU_B = 31,
  IN_PRIOR0_W = 32, IN_PRIOR0_B = 33, IN_PRIOR1_W = 34, IN_PRIOR1_B = 35,
  IN_PRIORLV_W = 36, IN_PRIORLV_B = 37, IN_PRIORMU_W = 38, IN_PRIORMU_B = 39,
  IN_Z0LV = 40, IN_Z0MU = 41
};

// --- workspace layout (bf16 element offsets unless noted)
constexpr size_t OFF_H1  = 0;                 // BL*128
constexpr size_t OFF_HU  = BL * 128;          // BL*128
constexpr size_t OFF_REP = 2 * BL * 128;
constexpr size_t R_G1F = OFF_REP;                       // [rk1;k1] 3x12 tiles
constexpr size_t R_G1B = R_G1F + 3 * 12 * 512;
constexpr size_t R_G2F = R_G1B + 3 * 12 * 512;          // [rk2;k2] 6x12 tiles
constexpr size_t R_G2B = R_G2F + 6 * 12 * 512;
constexpr size_t R_P1  = R_G2B + 6 * 12 * 512;          // prior W1 5x4
constexpr size_t R_Q1  = R_P1 + 5 * 4 * 512;            // post  W1 5x4
constexpr size_t R_P2  = R_Q1 + 5 * 4 * 512;            // prior W2 2x4
constexpr size_t R_Q2  = R_P2 + 2 * 4 * 512;
constexpr size_t R_PH  = R_Q2 + 2 * 4 * 512;            // prior heads 2x2
constexpr size_t R_QH  = R_PH + 2 * 2 * 512;
constexpr size_t R_D2  = R_QH + 2 * 2 * 512;            // dec W2 2x4
constexpr size_t REP_END = R_D2 + 2 * 4 * 512;
constexpr size_t OFF_PART_BYTES = ((REP_END * 2 + 255) / 256) * 256;   // 128*2 f32
constexpr size_t OFF_BIAS_BYTES = OFF_PART_BYTES + 1024;
// packed bias region (float offsets from bias base)
constexpr size_t B_G1F = 0, B_G1B = 256, B_G2F = 512, B_G2B = 768;
constexpr size_t B_PH = 1024, B_QH = 1056;    // 32 each
constexpr int SCAN_WGS = Bb / 32;             // 128

__device__ __forceinline__ float geluf(float x) {
  return 0.5f * x * (1.f + erff(x * 0.70710678118f));
}
__device__ __forceinline__ float sigmf(float x) {
  return 1.f / (1.f + __expf(-x));
}

// Async 16B global -> LDS copy (per-lane addresses), tracked by ASYNCcnt.
__device__ __forceinline__ void async_ld_b128(void* lds_dst, const void* gsrc) {
  uint32_t loff = (uint32_t)(uintptr_t)lds_dst;   // LDS aperture: low 32 = offset
  asm volatile("global_load_async_to_lds_b128 %0, %1, off"
               :: "v"(loff), "v"(gsrc) : "memory");
}
__device__ __forceinline__ void wait_async() {
  asm volatile("s_wait_asynccnt 0" ::: "memory");
}

// Load A fragment (16-bit, 16x32 tile) from LDS: lane<16 -> K {0..15},
// lane>=16 -> K {16..31}; base must already include (lane>>4)*16.
__device__ __forceinline__ v16bf load_a_frag(const __bf16* base) {
  v16bf a;
  const uint64_t* p0 = (const uint64_t*)base;
  const uint64_t* p1 = (const uint64_t*)(base + 8);
  uint64_t* d = (uint64_t*)&a;
  d[0] = p0[0]; d[1] = p0[1];
  d[2] = p1[0]; d[3] = p1[1];
  return a;
}

// Tiled GEMM: C[M x (ntc*16)] = bias + A[M x K] * Brep. A in LDS (row-major
// bf16, leading dim lda), Brep prepacked in WMMA B lane order. C fragment is
// initialized with an unconditional per-column bias broadcast.
// A K-offset per k-tile = ak0 + kk*32 ; B tile = (ktB0+kk)*NTtot + nt0+ntl.
__device__ __forceinline__ void gemm_tiles(
    const __bf16* __restrict__ A, int lda, int mtiles,
    const __bf16* __restrict__ Brep, int NTtot, int nt0, int ntc,
    int ktB0, int ktc, int ak0,
    float* __restrict__ C, int ldc,
    const float* __restrict__ bias, int boff)
{
  const int lane = threadIdx.x & 31;
  const int wave = threadIdx.x >> 5;
  const int m_in = lane >> 4;       // 0/1
  const int n_in = lane & 15;
  const int T = mtiles * ntc;
  for (int tt = wave; tt < T; tt += NWAVES) {
    const int mt = tt / ntc, ntl = tt % ntc;
    const float bv = bias[boff + ntl * 16 + n_in];
    v8f acc;
#pragma unroll
    for (int j = 0; j < 8; ++j) acc[j] = bv;
    const __bf16* arow = A + (size_t)(mt * 16 + n_in) * lda + ak0 + m_in * 16;
    for (int kk = 0; kk < ktc; ++kk) {
      v16bf a = load_a_frag(arow + kk * 32);
      v16bf b = *(const v16bf*)(Brep +
                 (((size_t)(ktB0 + kk) * NTtot + nt0 + ntl) * 32 + lane) * 16);
      acc = __builtin_amdgcn_wmma_f32_16x16x32_bf16(
          false, a, false, b, (short)0, acc, false, false);
    }
    const int r = mt * 16 + m_in * 8;
    float* cp = C + (size_t)r * ldc + ntl * 16 + n_in;
#pragma unroll
    for (int j = 0; j < 8; ++j) cp[(size_t)j * ldc] = acc[j];
  }
}

// ---------------------------------------------------------------------------
// Repack fp32 weight (K x N, row stride ldn) into WMMA B-fragment bf16 tiles.
// dst tile (kt,nt): lane l holds col n = nt*16+(l&15), rows k = kt*32 +
// (l>>4)*16 + i. Zero outside (Keff,N). s1/s2 (>=0): those dest k-rows are
// zero and higher rows read shifted source rows.
// ---------------------------------------------------------------------------
__global__ void k_repack(const float* __restrict__ W, int Keff, int N, int ldn,
                         int NT_local, int NT_total, int kt_off, int nt_off,
                         int s1, int s2, __bf16* __restrict__ dst)
{
  const int ktl = blockIdx.x / NT_local, ntl = blockIdx.x % NT_local;
  const int tid = threadIdx.x;            // 512
  const int l = tid >> 4, i = tid & 15;
  const int k = ktl * 32 + ((l >> 4) << 4) + i;
  const int n = ntl * 16 + (l & 15);
  float v = 0.f;
  if (k < Keff && n < N && k != s1 && k != s2) {
    const int ks = k - ((s1 >= 0 && k > s1) ? 1 : 0) - ((s2 >= 0 && k > s2) ? 1 : 0);
    v = W[(size_t)ks * ldn + n];
  }
  dst[(((size_t)(kt_off + ktl) * NT_total + (nt_off + ntl)) * 32 + l) * 16 + i] =
      (__bf16)v;
}

// Packed GRU bias: [0..127]=bi+br (z,r), [128..191]=br_h, [192..255]=bi_h.
__global__ void k_bias_gru(const float* __restrict__ bi,
                           const float* __restrict__ br,
                           float* __restrict__ dst)
{
  const int tid = threadIdx.x;            // 256
  if (tid < 128)      dst[tid] = bi[tid] + br[tid];
  else if (tid < 192) dst[tid] = br[tid];          // br[128..191]
  else                dst[tid] = bi[tid - 64];     // bi[128..191]
}

// Packed head bias: [0..5]=mu_b, [16..21]=lv_b, rest zero. 32 threads.
__global__ void k_bias_head(const float* __restrict__ mub,
                            const float* __restrict__ lvb,
                            float* __restrict__ dst)
{
  const int tid = threadIdx.x;
  float v = 0.f;
  if (tid < 6) v = mub[tid];
  else if (tid >= 16 && tid < 22) v = lvb[tid - 16];
  dst[tid] = v;
}

// ---------------------------------------------------------------------------
// GRU layer 1: A = [h(64) | u(6) | 0pad] K=96, B = stacked [rk ; k] (3x12
// tiles). Fully WMMA. blockIdx.y = direction, slab = 32 batch rows.
// ---------------------------------------------------------------------------
__global__ __launch_bounds__(256) void k_gru1(
    const float* __restrict__ u,
    const float* __restrict__ biasf, const float* __restrict__ biasb,
    const __bf16* __restrict__ repf, const __bf16* __restrict__ repb,
    __bf16* __restrict__ h1)
{
  const int dir = blockIdx.y;
  const float* bias = dir ? biasb : biasf;
  const __bf16* rep = dir ? repb : repf;
  const int b0 = blockIdx.x * 32;
  const int tid = threadIdx.x;

  __shared__ __align__(16) __bf16 aL[32 * 96];   // 0..63 h, 64..69 u, 70..95 pad
  __shared__ float comb[32 * 128];
  __shared__ float xh[32 * 64];
  __shared__ float rh[32 * 64];

  for (int e = tid; e < 32 * 96; e += 256) aL[e] = (__bf16)0.0f;
  __syncthreads();

  for (int t = 0; t < Lt; ++t) {
    const int tp = dir ? (Lt - 1 - t) : t;
    // phase A: u slab -> A cols 64..69
    for (int e = tid; e < 32 * DU; e += 256) {
      const int r = e / DU, d = e % DU;
      aL[r * 96 + 64 + d] = (__bf16)u[((size_t)(b0 + r) * Lt + tp) * DU + d];
    }
    __syncthreads();
    // phase B
    gemm_tiles(aL, 96, 2, rep, 12, 0, 8, 0, 3, 0,  comb, 128, bias, 0);
    gemm_tiles(aL, 96, 2, rep, 12, 8, 4, 0, 2, 0,  rh,   64,  bias, 128);
    gemm_tiles(aL, 96, 2, rep, 12, 8, 4, 2, 1, 64, xh,   64,  bias, 192);
    __syncthreads();
    // phase C: gates
    for (int e = tid; e < 32 * 64; e += 256) {
      const int r = e >> 6, i = e & 63;
      const float z  = sigmf(comb[r * 128 + i]);
      const float rg = sigmf(comb[r * 128 + 64 + i]);
      const float hhat = tanhf(xh[e] + rg * rh[e]);
      const float h = (float)aL[r * 96 + i];
      const float hn = z * h + (1.f - z) * hhat;
      aL[r * 96 + i] = (__bf16)hn;
      h1[(((size_t)(b0 + r) * Lt + tp) << 7) + dir * 64 + i] = (__bf16)hn;
    }
    __syncthreads();
  }
}

// ---------------------------------------------------------------------------
// GRU layer 2: A = [h(64) | x_t(128)] K=192, B = stacked [rk ; k] (6x12).
// x slab copied with async global->LDS b128.
// ---------------------------------------------------------------------------
__global__ __launch_bounds__(256) void k_gru2(
    const __bf16* __restrict__ h1,
    const float* __restrict__ biasf, const float* __restrict__ biasb,
    const __bf16* __restrict__ repf, const __bf16* __restrict__ repb,
    __bf16* __restrict__ hu)
{
  const int dir = blockIdx.y;
  const float* bias = dir ? biasb : biasf;
  const __bf16* rep = dir ? repb : repf;
  const int b0 = blockIdx.x * 32;
  const int tid = threadIdx.x;

  __shared__ __align__(16) __bf16 aL[32 * 192];   // 0..63 h, 64..191 x_t
  __shared__ float comb[32 * 128];
  __shared__ float xh[32 * 64];
  __shared__ float rh[32 * 64];

  for (int e = tid; e < 32 * 64; e += 256) aL[(e >> 6) * 192 + (e & 63)] = (__bf16)0.0f;
  __syncthreads();

  for (int t = 0; t < Lt; ++t) {
    const int tp = dir ? (Lt - 1 - t) : t;
    // phase A: async x slab (32 rows x 256B = 512 x 16B)
    for (int e = tid; e < 32 * 16; e += 256) {
      const int r = e >> 4, ch = e & 15;
      async_ld_b128(&aL[r * 192 + 64 + ch * 8],
                    h1 + (((size_t)(b0 + r) * Lt + tp) << 7) + ch * 8);
    }
    if (t + 1 < Lt) {   // prefetch next timestep's slab
      const int tpn = dir ? (Lt - 2 - t) : (t + 1);
      for (int r = tid; r < 32; r += 256)
        __builtin_prefetch(h1 + (((size_t)(b0 + r) * Lt + tpn) << 7), 0, 1);
    }
    wait_async();
    __syncthreads();
    // phase B
    gemm_tiles(aL, 192, 2, rep, 12, 0, 8, 0, 6, 0,  comb, 128, bias, 0);
    gemm_tiles(aL, 192, 2, rep, 12, 8, 4, 0, 2, 0,  rh,   64,  bias, 128);
    gemm_tiles(aL, 192, 2, rep, 12, 8, 4, 2, 4, 64, xh,   64,  bias, 192);
    __syncthreads();
    // phase C
    for (int e = tid; e < 32 * 64; e += 256) {
      const int r = e >> 6, i = e & 63;
      const float z  = sigmf(comb[r * 128 + i]);
      const float rg = sigmf(comb[r * 128 + 64 + i]);
      const float hhat = tanhf(xh[e] + rg * rh[e]);
      const float h = (float)aL[r * 192 + i];
      const float hn = z * h + (1.f - z) * hhat;
      aL[r * 192 + i] = (__bf16)hn;
      hu[(((size_t)(b0 + r) * Lt + tp) << 7) + dir * 64 + i] = (__bf16)hn;
    }
    __syncthreads();
  }
}

// ---------------------------------------------------------------------------
// Latent scan: unified A = [z(6) | y(1) | pad(1) | hu(128) | pad -> K=160].
// hu slab arrives via async b128 (16B-aligned at col 8). prior W1 repacked
// with rows 6,7 skipped; post W1 with row 7 skipped. Head biases are folded
// into the head GEMMs, so HOp/HOq hold final mu/lv (pre-clip).
// ---------------------------------------------------------------------------
__global__ __launch_bounds__(256) void k_scan(
    const float* __restrict__ y, const __bf16* __restrict__ hu,
    const __bf16* __restrict__ rp1, const __bf16* __restrict__ rp2,
    const __bf16* __restrict__ rpH,
    const __bf16* __restrict__ rq1, const __bf16* __restrict__ rq2,
    const __bf16* __restrict__ rqH,
    const __bf16* __restrict__ rd2,
    const float* __restrict__ pb1, const float* __restrict__ pb2,
    const float* __restrict__ qb1, const float* __restrict__ qb2,
    const float* __restrict__ bph, const float* __restrict__ bqh,
    const float* __restrict__ decW1, const float* __restrict__ decb1,
    const float* __restrict__ decb2,
    const float* __restrict__ meanW, const float* __restrict__ meanb,
    const float* __restrict__ z0mu,
    float* __restrict__ yhat, float* __restrict__ partials)
{
  const int b0 = blockIdx.x * 32;
  const int tid = threadIdx.x;

  __shared__ __align__(16) __bf16 Abuf[32 * 160];
  __shared__ float Pbuf[32 * 64];
  __shared__ __align__(16) __bf16 Ha[32 * 64];
  __shared__ __align__(16) __bf16 Hh[32 * 64];
  __shared__ float HOp[32 * 32];
  __shared__ float HOq[32 * 32];
  __shared__ float dotb[256];
  __shared__ float redv[2];

  for (int e = tid; e < 32 * 160; e += 256) {
    const int c = e % 160;
    Abuf[e] = (__bf16)((c < DZ) ? z0mu[c] : 0.f);
  }
  if (tid < 2) redv[tid] = 0.f;
  float recs = 0.f, kls = 0.f;
  __syncthreads();

  for (int t = 0; t < Lt; ++t) {
    // slab: hu_t -> cols 8..135 (async), y_t -> col 6
    for (int e = tid; e < 32 * 16; e += 256) {
      const int r = e >> 4, ch = e & 15;
      async_ld_b128(&Abuf[r * 160 + 8 + ch * 8],
                    hu + (((size_t)(b0 + r) * Lt + t) << 7) + ch * 8);
    }
    for (int r = tid; r < 32; r += 256) {
      Abuf[r * 160 + 6] = (__bf16)y[(size_t)(b0 + r) * Lt + t];
      if (t + 1 < Lt)
        __builtin_prefetch(hu + (((size_t)(b0 + r) * Lt + t + 1) << 7), 0, 1);
    }
    wait_async();
    __syncthreads();

    // ---- prior MLP
    gemm_tiles(Abuf, 160, 2, rp1, 4, 0, 4, 0, 5, 0, Pbuf, 64, pb1, 0);
    __syncthreads();
    for (int e = tid; e < 32 * 64; e += 256) Ha[e] = (__bf16)geluf(Pbuf[e]);
    __syncthreads();
    gemm_tiles(Ha, 64, 2, rp2, 4, 0, 4, 0, 2, 0, Pbuf, 64, pb2, 0);
    __syncthreads();
    for (int e = tid; e < 32 * 64; e += 256) Hh[e] = (__bf16)geluf(Pbuf[e]);
    __syncthreads();
    gemm_tiles(Hh, 64, 2, rpH, 2, 0, 2, 0, 2, 0, HOp, 32, bph, 0);
    __syncthreads();

    // ---- posterior MLP
    gemm_tiles(Abuf, 160, 2, rq1, 4, 0, 4, 0, 5, 0, Pbuf, 64, qb1, 0);
    __syncthreads();
    for (int e = tid; e < 32 * 64; e += 256) Ha[e] = (__bf16)geluf(Pbuf[e]);
    __syncthreads();
    gemm_tiles(Ha, 64, 2, rq2, 4, 0, 4, 0, 2, 0, Pbuf, 64, qb2, 0);
    __syncthreads();
    for (int e = tid; e < 32 * 64; e += 256) Hh[e] = (__bf16)geluf(Pbuf[e]);
    __syncthreads();
    gemm_tiles(Hh, 64, 2, rqH, 2, 0, 2, 0, 2, 0, HOq, 32, bqh, 0);
    __syncthreads();

    // ---- decoder layer 1 (K=6, VALU) + z_t update  (z_t = mu_q = HOq[.,0..5])
    for (int e = tid; e < 32 * 64; e += 256) {
      const int r = e >> 6, c = e & 63;
      float s = decb1[c];
#pragma unroll
      for (int d = 0; d < DZ; ++d)
        s += HOq[r * 32 + d] * decW1[d * 64 + c];
      Ha[e] = (__bf16)geluf(s);
    }
    for (int e = tid; e < 32 * DZ; e += 256) {
      const int r = e / DZ, d = e % DZ;
      Abuf[r * 160 + d] = (__bf16)HOq[r * 32 + d];
    }
    __syncthreads();
    // decoder layer 2
    gemm_tiles(Ha, 64, 2, rd2, 4, 0, 4, 0, 2, 0, Pbuf, 64, decb2, 0);
    __syncthreads();
    for (int e = tid; e < 32 * 64; e += 256) Pbuf[e] = geluf(Pbuf[e]);
    __syncthreads();

    // ---- yhat dot (8 threads per row), then rec/kl per row
    {
      const int g = tid & 7, r = tid >> 3;
      float s = 0.f;
#pragma unroll
      for (int c = g * 8; c < g * 8 + 8; ++c) s += Pbuf[r * 64 + c] * meanW[c];
      dotb[tid] = s;
    }
    __syncthreads();
    if (tid < 32) {
      const int r = tid;
      float yh = meanb[0];
#pragma unroll
      for (int g = 0; g < 8; ++g) yh += dotb[r * 8 + g];
      const float yt = y[(size_t)(b0 + r) * Lt + t];
      yhat[(size_t)(b0 + r) * Lt + t] = yh;
      recs += (yt - yh) * (yt - yh);
      float kl = 0.f;
#pragma unroll
      for (int d = 0; d < DZ; ++d) {
        const float mup = HOp[r * 32 + d];
        const float muq = HOq[r * 32 + d];
        const float lvp = fminf(fmaxf(HOp[r * 32 + 16 + d], -8.f), 8.f);
        const float lvq = fminf(fmaxf(HOq[r * 32 + 16 + d], -8.f), 8.f);
        const float dm = muq - mup;
        kl += lvp - lvq + (__expf(lvq) + dm * dm) / (__expf(lvp) + 1e-8f) - 1.f;
      }
      kls += 0.5f * kl;
    }
    __syncthreads();
  }

  atomicAdd(&redv[0], recs);
  atomicAdd(&redv[1], kls);
  __syncthreads();
  if (tid == 0) {
    partials[blockIdx.x * 2 + 0] = redv[0];
    partials[blockIdx.x * 2 + 1] = redv[1];
  }
}

__global__ void k_loss(const float* __restrict__ partials, int nwg,
                       float* __restrict__ out)
{
  __shared__ float sr[256], sk[256];
  const int tid = threadIdx.x;
  float r = 0.f, k = 0.f;
  for (int i = tid; i < nwg; i += 256) { r += partials[2 * i]; k += partials[2 * i + 1]; }
  sr[tid] = r; sk[tid] = k;
  __syncthreads();
  for (int s = 128; s > 0; s >>= 1) {
    if (tid < s) { sr[tid] += sr[tid + s]; sk[tid] += sk[tid + s]; }
    __syncthreads();
  }
  if (tid == 0) {
    const float inv = 1.f / (float)(4096.0 * 128.0);
    out[0] = 0.5f * sr[0] * inv + 0.01f * sk[0] * inv;
  }
}

// ---------------------------------------------------------------------------
extern "C" void kernel_launch(void* const* d_in, const int* in_sizes, int n_in,
                              void* d_out, int out_size, void* d_ws, size_t ws_size,
                              hipStream_t stream)
{
  (void)in_sizes; (void)n_in; (void)out_size; (void)ws_size;
  const float* Y = (const float*)d_in[IN_Y];
  const float* U = (const float*)d_in[IN_U];
  auto F = [&](int i) { return (const float*)d_in[i]; };

  __bf16* ws = (__bf16*)d_ws;
  __bf16* h1 = ws + OFF_H1;
  __bf16* hu = ws + OFF_HU;
  float* partials = (float*)((char*)d_ws + OFF_PART_BYTES);
  float* biasws = (float*)((char*)d_ws + OFF_BIAS_BYTES);
  float* yhat = (float*)d_out;
  float* lossp = yhat + BL;

  // ---- weight repacking into WMMA B-lane layout (bf16)
  // gru1 stacked [rk (64x192) ; k (6x192)]: kt 0..1 from rk, kt 2 from k
  k_repack<<<2 * 12, 512, 0, stream>>>(F(IN_G1F_RK), 64, 192, 192, 12, 12, 0, 0, -1, -1, ws + R_G1F);
  k_repack<<<1 * 12, 512, 0, stream>>>(F(IN_G1F_K),   6, 192, 192, 12, 12, 2, 0, -1, -1, ws + R_G1F);
  k_repack<<<2 * 12, 512, 0, stream>>>(F(IN_G1B_RK), 64, 192, 192, 12, 12, 0, 0, -1, -1, ws + R_G1B);
  k_repack<<<1 * 12, 512, 0, stream>>>(F(IN_G1B_K),   6, 192, 192, 12, 12, 2, 0, -1, -1, ws + R_G1B);
  // gru2 stacked [rk (64x192) ; k (128x192)]: kt 0..1 from rk, 2..5 from k
  k_repack<<<2 * 12, 512, 0, stream>>>(F(IN_G2F_RK), 64, 192, 192, 12, 12, 0, 0, -1, -1, ws + R_G2F);
  k_repack<<<4 * 12, 512, 0, stream>>>(F(IN_G2F_K), 128, 192, 192, 12, 12, 2, 0, -1, -1, ws + R_G2F);
  k_repack<<<2 * 12, 512, 0, stream>>>(F(IN_G2B_RK), 64, 192, 192, 12, 12, 0, 0, -1, -1, ws + R_G2B);
  k_repack<<<4 * 12, 512, 0, stream>>>(F(IN_G2B_K), 128, 192, 192, 12, 12, 2, 0, -1, -1, ws + R_G2B);
  // prior W1 (134x64) into K=160 [z|y|pad|hu]: rows 6 (y) and 7 (pad) skipped
  k_repack<<<5 * 4, 512, 0, stream>>>(F(IN_PRIOR0_W), 136, 64, 64, 4, 4, 0, 0, 6, 7, ws + R_P1);
  // post W1 (135x64): row 7 (pad) skipped
  k_repack<<<5 * 4, 512, 0, stream>>>(F(IN_POST0_W), 136, 64, 64, 4, 4, 0, 0, 7, -1, ws + R_Q1);
  k_repack<<<2 * 4, 512, 0, stream>>>(F(IN_PRIOR1_W), 64, 64, 64, 4, 4, 0, 0, -1, -1, ws + R_P2);
  k_repack<<<2 * 4, 512, 0, stream>>>(F(IN_POST1_W), 64, 64, 64, 4, 4, 0, 0, -1, -1, ws + R_Q2);
  // heads: [mu | lv] each padded 6->16, combined NT_total=2
  k_repack<<<2 * 1, 512, 0, stream>>>(F(IN_PRIORMU_W), 64, 6, 6, 1, 2, 0, 0, -1, -1, ws + R_PH);
  k_repack<<<2 * 1, 512, 0, stream>>>(F(IN_PRIORLV_W), 64, 6, 6, 1, 2, 0, 1, -1, -1, ws + R_PH);
  k_repack<<<2 * 1, 512, 0, stream>>>(F(IN_POSTMU_W), 64, 6, 6, 1, 2, 0, 0, -1, -1, ws + R_QH);
  k_repack<<<2 * 1, 512, 0, stream>>>(F(IN_POSTLV_W), 64, 6, 6, 1, 2, 0, 1, -1, -1, ws + R_QH);
  k_repack<<<2 * 4, 512, 0, stream>>>(F(IN_DEC1_W), 64, 64, 64, 4, 4, 0, 0, -1, -1, ws + R_D2);

  // ---- packed bias vectors
  k_bias_gru<<<1, 256, 0, stream>>>(F(IN_G1F_BI), F(IN_G1F_BR), biasws + B_G1F);
  k_bias_gru<<<1, 256, 0, stream>>>(F(IN_G1B_BI), F(IN_G1B_BR), biasws + B_G1B);
  k_bias_gru<<<1, 256, 0, stream>>>(F(IN_G2F_BI), F(IN_G2F_BR), biasws + B_G2F);
  k_bias_gru<<<1, 256, 0, stream>>>(F(IN_G2B_BI), F(IN_G2B_BR), biasws + B_G2B);
  k_bias_head<<<1, 32, 0, stream>>>(F(IN_PRIORMU_B), F(IN_PRIORLV_B), biasws + B_PH);
  k_bias_head<<<1, 32, 0, stream>>>(F(IN_POSTMU_B), F(IN_POSTLV_B), biasws + B_QH);

  // ---- GRU stack
  k_gru1<<<dim3(Bb / 32, 2), 256, 0, stream>>>(
      U, biasws + B_G1F, biasws + B_G1B, ws + R_G1F, ws + R_G1B, h1);
  k_gru2<<<dim3(Bb / 32, 2), 256, 0, stream>>>(
      h1, biasws + B_G2F, biasws + B_G2B, ws + R_G2F, ws + R_G2B, hu);

  // ---- latent scan
  k_scan<<<SCAN_WGS, 256, 0, stream>>>(
      Y, hu,
      ws + R_P1, ws + R_P2, ws + R_PH,
      ws + R_Q1, ws + R_Q2, ws + R_QH,
      ws + R_D2,
      F(IN_PRIOR0_B), F(IN_PRIOR1_B),
      F(IN_POST0_B), F(IN_POST1_B),
      biasws + B_PH, biasws + B_QH,
      F(IN_DEC0_W), F(IN_DEC0_B), F(IN_DEC1_B),
      F(IN_DECMEAN_W), F(IN_DECMEAN_B),
      F(IN_Z0MU),
      yhat, partials);

  // ---- loss
  k_loss<<<1, 256, 0, stream>>>(partials, SCAN_WGS, lossp);
}
